// InputFeedRNNDecoder_88991722373680
// MI455X (gfx1250) — compile-verified, matrix operand order checked by reading, not run on hardware
//
#include <hip/hip_runtime.h>
#include <hip/hip_bf16.h>
#include <math.h>

// InputFeedRNNDecoder for MI455X (gfx1250, wave32).
// Strategy:
//  - Down-convert memory_bank / inputs / weights to bf16 once (workspace).
//    proj_mem(bf16) + memory_bank(bf16) = 128 MB -> resident in the 192 MB L2,
//    which is what the per-step attention GEMVs stream every timestep.
//  - All GEMMs (proj_mem precompute, LSTM gates, output projection) use
//    v_wmma_f32_16x16x32_bf16 (f32 accumulate).
//  - proj GEMM stages its 16-row A panel into LDS via the Tensor Data Mover
//    (TENSOR_LOAD_TO_LDS + s_wait_tensorcnt), then 8 waves sweep 32 output
//    tiles from LDS: HBM -> TDM -> LDS -> WMMA, with 32x A reuse.
//  - Attention score/context are GEMV-shaped -> coalesced VALU + wave32
//    shuffle reductions; softmax uses LDS tree reduction.

#define B_ 64
#define T_ 128
#define S_ 1024
#define E_ 512
#define H_ 512

typedef __attribute__((ext_vector_type(16))) __bf16 v16bf;
typedef __attribute__((ext_vector_type(8)))  __bf16 v8bf;
typedef __attribute__((ext_vector_type(8)))  float  v8f;
typedef __attribute__((ext_vector_type(4)))  unsigned int v4u;
typedef __attribute__((ext_vector_type(8)))  int v8i;
typedef __attribute__((ext_vector_type(4)))  int v4i;

static __device__ __forceinline__ float bf2f(__bf16 b) {
  unsigned short s = __builtin_bit_cast(unsigned short, b);
  unsigned u = ((unsigned)s) << 16;
  return __builtin_bit_cast(float, u);
}
static __device__ __forceinline__ __bf16 f2bf(float f) {
  unsigned u = __builtin_bit_cast(unsigned, f);
  unsigned r = u + 0x7FFFu + ((u >> 16) & 1u);  // round-to-nearest-even
  unsigned short s = (unsigned short)(r >> 16);
  return __builtin_bit_cast(__bf16, s);
}

// ---- WMMA helpers -----------------------------------------------------------
// A-matrix 16x32 bf16 fragment per ISA 7.12.2: lane<16 -> row M=lane, holds
// K = {0..7, 16..23}; lane>=16 -> row M=lane-16, K = {8..15, 24..31}.
// Loaded as two 16B vectors (global_load_b128 / ds_load_b128).
static __device__ __forceinline__ v16bf load_frag(const __bf16* __restrict__ base,
                                                  int ld, int lane) {
  const int g = (lane >> 4) & 1;
  const int m = lane & 15;
  const __bf16* q = base + (size_t)m * ld + g * 8;
  v8bf lo = *reinterpret_cast<const v8bf*>(q);        // K = g*8 + 0..7
  v8bf hi = *reinterpret_cast<const v8bf*>(q + 16);   // K = g*8 + 16..23
  return __builtin_shufflevector(lo, hi, 0, 1, 2, 3, 4, 5, 6, 7,
                                 8, 9, 10, 11, 12, 13, 14, 15);
}

static __device__ __forceinline__ v8f wmma_bf16(v16bf a, v16bf b, v8f c) {
  // 8 args: (neg_a, A, neg_b, B, c_mod, C, reuse_a, reuse_b)
  return __builtin_amdgcn_wmma_f32_16x16x32_bf16(false, a, false, b,
                                                 (short)0, c, false, false);
}

// ---- conversion -------------------------------------------------------------
__global__ void k_f32_to_bf16(const float* __restrict__ src,
                              __bf16* __restrict__ dst, int n) {
  int i = blockIdx.x * blockDim.x + threadIdx.x;
  if (i < n) dst[i] = f2bf(src[i]);
}

// ---- proj_mem = memory_bank @ W_a^T  (bf16 in, bf16 out, f32 acc) ----------
// Block = 8 waves. Each block owns one 16-row A panel (16x512 bf16 = 16 KB),
// staged into LDS once via the Tensor Data Mover, then 8 waves compute the 32
// ht output tiles from LDS. Grid = B*(S/16) = 4096 blocks.
__global__ void k_proj_gemm(const __bf16* __restrict__ mem,
                            const __bf16* __restrict__ Wa,
                            __bf16* __restrict__ proj) {
  __shared__ __align__(16) __bf16 atile[16 * H_];  // 16 KB
  const int lane = threadIdx.x & 31;
  const int wave = threadIdx.x >> 5;
  const int b  = blockIdx.x / (S_ / 16);
  const int st = blockIdx.x % (S_ / 16);
  const __bf16* gsrc = mem + ((size_t)b * S_ + st * 16) * H_;

#if __has_builtin(__builtin_amdgcn_tensor_load_to_lds) && \
    __has_builtin(__builtin_amdgcn_s_wait_tensorcnt)
  if (wave == 0) {
    // D# per ISA 8.3/8.4: 2D tile, data_size=2B, 512x16, stride 512.
    unsigned long long ga = (unsigned long long)(uintptr_t)gsrc;
    unsigned lds = (unsigned)(uintptr_t)(&atile[0]);
    v4u g0;
    g0[0] = 1u;                                        // count=1 (valid user D#)
    g0[1] = lds;                                       // lds_addr
    g0[2] = (unsigned)(ga & 0xFFFFFFFFu);              // global_addr[31:0]
    g0[3] = (unsigned)((ga >> 32) & 0x1FFFFFFu)        // global_addr[56:32]
            | (2u << 30);                              // type = 2 ("image")
    v8i g1;
    g1[0] = (1 << 16);       // workgroup_mask=0, data_size=1 -> 2 bytes
    g1[1] = (512 << 16);     // tensor_dim0 = 512 (bits 79:48, low half)
    g1[2] = (16 << 16);      // tensor_dim1 = 16  (bits 111:80, low half)
    g1[3] = (512 << 16);     // tile_dim0 = 512   (bits 127:112)
    g1[4] = 16;              // tile_dim1 = 16, tile_dim2 = 0
    g1[5] = 512;             // tensor_dim0_stride = 512 (bits 207:160)
    g1[6] = 0;
    g1[7] = 0;
    v4i z4 = {};
#if __clang_major__ >= 23
    v8i z8 = {};
    __builtin_amdgcn_tensor_load_to_lds(g0, g1, z4, z4, z8, 0);
#else
    __builtin_amdgcn_tensor_load_to_lds(g0, g1, z4, z4, 0);
#endif
    __builtin_amdgcn_s_wait_tensorcnt(0);
  }
  __syncthreads();
#else
  // Fallback: cooperative 16B-vector copy global -> LDS.
  for (int i = threadIdx.x * 8; i < 16 * H_; i += 256 * 8)
    *reinterpret_cast<v8bf*>(&atile[i]) =
        *reinterpret_cast<const v8bf*>(gsrc + i);
  __syncthreads();
#endif

  for (int ht = wave; ht < H_ / 16; ht += 8) {
    const __bf16* Bbase = Wa + (size_t)(ht * 16) * H_;  // W_a[n][k], ld=H_
    v8f acc = {};
#pragma unroll 4
    for (int k0 = 0; k0 < H_; k0 += 32) {
      v16bf a  = load_frag(&atile[k0], H_, lane);       // ds_load_b128 path
      v16bf bb = load_frag(Bbase + k0, H_, lane);
      acc = wmma_bf16(a, bb, acc);
    }
    __bf16* outp = proj + ((size_t)b * S_ + st * 16) * H_ + ht * 16;
    const int g = lane >> 4, n = lane & 15;
#pragma unroll
    for (int rr = 0; rr < 8; ++rr)
      outp[(size_t)(rr + g * 8) * H_ + n] = f2bf(acc[rr]);
  }
}

// ---- gates = [x_t | feed] @ W_ih^T + h @ W_hh^T  (f32 out) ------------------
// Tiles = (B/16)*(4H/16) = 4*128 = 512; 8 waves/block -> grid 64.
__global__ void k_gates_gemm(const __bf16* __restrict__ xbf,
                             const __bf16* __restrict__ feedbf,
                             const __bf16* __restrict__ hbf,
                             const __bf16* __restrict__ Wih,
                             const __bf16* __restrict__ Whh,
                             float* __restrict__ gates, int t) {
  const int lane = threadIdx.x & 31;
  const int wave = threadIdx.x >> 5;
  const int tile = blockIdx.x * 8 + wave;
  const int NT = (4 * H_) / 16;  // 128
  const int bt = tile / NT;
  const int nt = tile % NT;
  const int ldW = E_ + H_;  // 1024

  v8f acc = {};
  // x segment: A row m = batch bt*16+m, x[(b*T+t)*E + k]; row stride T*E
  const __bf16* Ax = xbf + ((size_t)(bt * 16) * T_ + t) * E_;
  const __bf16* Wx = Wih + (size_t)(nt * 16) * ldW;
#pragma unroll 4
  for (int k0 = 0; k0 < E_; k0 += 32)
    acc = wmma_bf16(load_frag(Ax + k0, T_ * E_, lane),
                    load_frag(Wx + k0, ldW, lane), acc);
  // feed segment
  const __bf16* Af = feedbf + (size_t)(bt * 16) * H_;
  const __bf16* Wf = Wx + E_;
#pragma unroll 4
  for (int k0 = 0; k0 < H_; k0 += 32)
    acc = wmma_bf16(load_frag(Af + k0, H_, lane),
                    load_frag(Wf + k0, ldW, lane), acc);
  // recurrent segment
  const __bf16* Ah = hbf + (size_t)(bt * 16) * H_;
  const __bf16* Wh = Whh + (size_t)(nt * 16) * H_;
#pragma unroll 4
  for (int k0 = 0; k0 < H_; k0 += 32)
    acc = wmma_bf16(load_frag(Ah + k0, H_, lane),
                    load_frag(Wh + k0, H_, lane), acc);

  float* out = gates + (size_t)(bt * 16) * (4 * H_) + nt * 16;
  const int g = lane >> 4, n = lane & 15;
#pragma unroll
  for (int rr = 0; rr < 8; ++rr)
    out[(size_t)(rr + g * 8) * (4 * H_) + n] = acc[rr];
}

// ---- LSTM elementwise -------------------------------------------------------
__global__ void k_lstm(const float* __restrict__ gates,
                       const float* __restrict__ bias,
                       float* __restrict__ c, float* __restrict__ h,
                       __bf16* __restrict__ hbf,
                       float* __restrict__ out_rnn,
                       float* __restrict__ out_hT, float* __restrict__ out_cT,
                       int t) {
  int i = blockIdx.x * blockDim.x + threadIdx.x;
  if (i >= B_ * H_) return;
  const int b = i >> 9, n = i & (H_ - 1);
  const float* g = gates + (size_t)b * (4 * H_);
  float gi = g[n]            + bias[n];
  float gf = g[H_ + n]       + bias[H_ + n];
  float gg = g[2 * H_ + n]   + bias[2 * H_ + n];
  float go = g[3 * H_ + n]   + bias[3 * H_ + n];
  float si = 1.f / (1.f + expf(-gi));
  float sf = 1.f / (1.f + expf(-gf));
  float so = 1.f / (1.f + expf(-go));
  float cn = sf * c[i] + si * tanhf(gg);
  float hn = so * tanhf(cn);
  c[i] = cn; h[i] = hn; hbf[i] = f2bf(hn);
  out_rnn[((size_t)b * T_ + t) * H_ + n] = hn;
  out_hT[i] = hn; out_cT[i] = cn;  // last step's values survive
}

// ---- attention scores: scores[b,s] = h[b,:]·proj[b,s,:] (wave per s) -------
__global__ void k_scores(const float* __restrict__ h,
                         const __bf16* __restrict__ proj,
                         const int* __restrict__ mask,
                         float* __restrict__ scores) {
  const int lane = threadIdx.x & 31;
  const int wid = (blockIdx.x * blockDim.x + threadIdx.x) >> 5;  // b*S + s
  const int b = wid >> 10, s = wid & (S_ - 1);
  const float* hv = h + (size_t)b * H_;
  const __bf16* p = proj + ((size_t)b * S_ + s) * H_;
  float acc = 0.f;
#pragma unroll 4
  for (int k = lane; k < H_; k += 32) acc += hv[k] * bf2f(p[k]);
#pragma unroll
  for (int off = 16; off > 0; off >>= 1) acc += __shfl_xor(acc, off, 32);
  if (lane == 0)
    scores[wid] = mask[b * S_ + s] ? acc : -1e9f;
}

// ---- softmax over S (block per batch) --------------------------------------
__global__ void k_softmax(const float* __restrict__ scores,
                          float* __restrict__ attn,
                          float* __restrict__ out_attn, int t) {
  __shared__ float red[256];
  const int b = blockIdx.x, tid = threadIdx.x;
  const float* sc = scores + (size_t)b * S_;
  float m = -3.4e38f;
  for (int s = tid; s < S_; s += 256) m = fmaxf(m, sc[s]);
  red[tid] = m; __syncthreads();
  for (int off = 128; off > 0; off >>= 1) {
    if (tid < off) red[tid] = fmaxf(red[tid], red[tid + off]);
    __syncthreads();
  }
  m = red[0]; __syncthreads();
  float sum = 0.f;
  for (int s = tid; s < S_; s += 256) sum += expf(sc[s] - m);
  red[tid] = sum; __syncthreads();
  for (int off = 128; off > 0; off >>= 1) {
    if (tid < off) red[tid] += red[tid + off];
    __syncthreads();
  }
  const float inv = 1.f / red[0];
  for (int s = tid; s < S_; s += 256) {
    float a = expf(sc[s] - m) * inv;
    attn[(size_t)b * S_ + s] = a;
    out_attn[((size_t)b * T_ + t) * S_ + s] = a;
  }
}

// ---- context: ctx[b,d] = sum_s attn[b,s]*mem[b,s,d] ------------------------
__global__ void k_ctx(const float* __restrict__ attn,
                      const __bf16* __restrict__ mem,
                      float* __restrict__ ctx, __bf16* __restrict__ ctxbf) {
  const int b = blockIdx.x >> 1;
  const int d = ((blockIdx.x & 1) * 256) + threadIdx.x;
  const float* a = attn + (size_t)b * S_;
  const __bf16* m = mem + (size_t)b * S_ * H_ + d;
  float acc = 0.f;
  for (int s = 0; s < S_; ++s) {
    if ((s & 63) == 0)
      __builtin_prefetch(m + (size_t)(s + 64) * H_, 0, 0);  // global_prefetch_b8
    acc += a[s] * bf2f(m[(size_t)s * H_]);
  }
  ctx[(size_t)b * H_ + d] = acc;
  ctxbf[(size_t)b * H_ + d] = f2bf(acc);
}

// ---- out = tanh([ctx|h] @ W_c^T); also next input_feed ----------------------
// Tiles = (B/16)*(H/16) = 128; 8 waves/block -> grid 16.
__global__ void k_out_gemm(const __bf16* __restrict__ ctxbf,
                           const __bf16* __restrict__ hbf,
                           const __bf16* __restrict__ Wc,
                           float* __restrict__ out_dec,
                           float* __restrict__ feed, __bf16* __restrict__ feedbf,
                           float* __restrict__ out_fT, int t) {
  const int lane = threadIdx.x & 31;
  const int wave = threadIdx.x >> 5;
  const int tile = blockIdx.x * 8 + wave;
  const int NT = H_ / 16;  // 32
  const int bt = tile / NT;
  const int nt = tile % NT;
  const int ldW = 2 * H_;  // 1024

  v8f acc = {};
  const __bf16* Ac = ctxbf + (size_t)(bt * 16) * H_;
  const __bf16* W0 = Wc + (size_t)(nt * 16) * ldW;
#pragma unroll 4
  for (int k0 = 0; k0 < H_; k0 += 32)
    acc = wmma_bf16(load_frag(Ac + k0, H_, lane),
                    load_frag(W0 + k0, ldW, lane), acc);
  const __bf16* Ah = hbf + (size_t)(bt * 16) * H_;
  const __bf16* W1 = W0 + H_;
#pragma unroll 4
  for (int k0 = 0; k0 < H_; k0 += 32)
    acc = wmma_bf16(load_frag(Ah + k0, H_, lane),
                    load_frag(W1 + k0, ldW, lane), acc);

  const int g = lane >> 4, n = lane & 15;
#pragma unroll
  for (int rr = 0; rr < 8; ++rr) {
    const int m = bt * 16 + rr + g * 8;
    const int hh = nt * 16 + n;
    float v = tanhf(acc[rr]);
    out_dec[((size_t)m * T_ + t) * H_ + hh] = v;
    feed[(size_t)m * H_ + hh] = v;
    feedbf[(size_t)m * H_ + hh] = f2bf(v);
    out_fT[(size_t)m * H_ + hh] = v;
  }
}

// ---- state init -------------------------------------------------------------
__global__ void k_init(const float* __restrict__ h0, const float* __restrict__ c0,
                       float* __restrict__ h, float* __restrict__ c,
                       float* __restrict__ feed,
                       __bf16* __restrict__ hbf, __bf16* __restrict__ feedbf) {
  int i = blockIdx.x * blockDim.x + threadIdx.x;
  if (i >= B_ * H_) return;
  h[i] = h0[i]; c[i] = c0[i]; feed[i] = 0.f;
  hbf[i] = f2bf(h0[i]); feedbf[i] = f2bf(0.f);
}

// ---- host launcher ----------------------------------------------------------
extern "C" void kernel_launch(void* const* d_in, const int* in_sizes, int n_in,
                              void* d_out, int out_size, void* d_ws, size_t ws_size,
                              hipStream_t stream) {
  (void)in_sizes; (void)n_in; (void)out_size; (void)ws_size;
  const float* inputs = (const float*)d_in[0];
  const float* memory = (const float*)d_in[1];
  const int*   mask   = (const int*)d_in[2];
  const float* h0     = (const float*)d_in[3];
  const float* c0     = (const float*)d_in[4];
  const float* W_ih   = (const float*)d_in[5];
  const float* W_hh   = (const float*)d_in[6];
  const float* bias   = (const float*)d_in[7];
  const float* W_a    = (const float*)d_in[8];
  const float* W_c    = (const float*)d_in[9];
  float* out = (float*)d_out;

  // workspace partition (256B aligned slices)
  char* p = (char*)d_ws;
  auto take = [&](size_t bytes) -> char* {
    char* r = p; p += (bytes + 255) & ~(size_t)255; return r;
  };
  __bf16* mem_bf  = (__bf16*)take((size_t)B_ * S_ * H_ * 2);
  __bf16* proj_bf = (__bf16*)take((size_t)B_ * S_ * H_ * 2);
  __bf16* x_bf    = (__bf16*)take((size_t)B_ * T_ * E_ * 2);
  __bf16* Wih_bf  = (__bf16*)take((size_t)4 * H_ * (E_ + H_) * 2);
  __bf16* Whh_bf  = (__bf16*)take((size_t)4 * H_ * H_ * 2);
  __bf16* Wa_bf   = (__bf16*)take((size_t)H_ * H_ * 2);
  __bf16* Wc_bf   = (__bf16*)take((size_t)H_ * 2 * H_ * 2);
  float*  hbuf    = (float*)take((size_t)B_ * H_ * 4);
  float*  cbuf    = (float*)take((size_t)B_ * H_ * 4);
  float*  feedbuf = (float*)take((size_t)B_ * H_ * 4);
  float*  ctxbuf  = (float*)take((size_t)B_ * H_ * 4);
  __bf16* h_bf    = (__bf16*)take((size_t)B_ * H_ * 2);
  __bf16* feed_bf = (__bf16*)take((size_t)B_ * H_ * 2);
  __bf16* ctx_bf  = (__bf16*)take((size_t)B_ * H_ * 2);
  float*  gates   = (float*)take((size_t)B_ * 4 * H_ * 4);
  float*  scores  = (float*)take((size_t)B_ * S_ * 4);
  float*  attn    = (float*)take((size_t)B_ * S_ * 4);

  // output partition
  float* out_dec  = out;
  float* out_rnn  = out + (size_t)B_ * T_ * H_;
  float* out_attn = out + 2 * (size_t)B_ * T_ * H_;
  float* out_hT   = out_attn + (size_t)B_ * T_ * S_;
  float* out_cT   = out_hT + (size_t)B_ * H_;
  float* out_fT   = out_cT + (size_t)B_ * H_;

  // one-time down-conversions
  int n;
  n = B_ * S_ * H_;
  k_f32_to_bf16<<<(n + 255) / 256, 256, 0, stream>>>(memory, mem_bf, n);
  n = B_ * T_ * E_;
  k_f32_to_bf16<<<(n + 255) / 256, 256, 0, stream>>>(inputs, x_bf, n);
  n = 4 * H_ * (E_ + H_);
  k_f32_to_bf16<<<(n + 255) / 256, 256, 0, stream>>>(W_ih, Wih_bf, n);
  n = 4 * H_ * H_;
  k_f32_to_bf16<<<(n + 255) / 256, 256, 0, stream>>>(W_hh, Whh_bf, n);
  n = H_ * H_;
  k_f32_to_bf16<<<(n + 255) / 256, 256, 0, stream>>>(W_a, Wa_bf, n);
  n = H_ * 2 * H_;
  k_f32_to_bf16<<<(n + 255) / 256, 256, 0, stream>>>(W_c, Wc_bf, n);

  // proj_mem precompute (TDM-staged WMMA GEMM); grid = B*(S/16) panels
  k_proj_gemm<<<B_ * (S_ / 16), 256, 0, stream>>>(mem_bf, Wa_bf, proj_bf);

  // state init
  k_init<<<(B_ * H_ + 255) / 256, 256, 0, stream>>>(h0, c0, hbuf, cbuf, feedbuf,
                                                    h_bf, feed_bf);

  // sequential decode
  for (int t = 0; t < T_; ++t) {
    k_gates_gemm<<<64, 256, 0, stream>>>(x_bf, feed_bf, h_bf, Wih_bf, Whh_bf, gates, t);
    k_lstm<<<(B_ * H_ + 255) / 256, 256, 0, stream>>>(gates, bias, cbuf, hbuf, h_bf,
                                                      out_rnn, out_hT, out_cT, t);
    k_scores<<<(B_ * S_) / 8, 256, 0, stream>>>(hbuf, proj_bf, mask, scores);
    k_softmax<<<B_, 256, 0, stream>>>(scores, attn, out_attn, t);
    k_ctx<<<B_ * 2, 256, 0, stream>>>(attn, mem_bf, ctxbuf, ctx_bf);
    k_out_gemm<<<16, 256, 0, stream>>>(ctx_bf, h_bf, Wc_bf, out_dec, feedbuf,
                                       feed_bf, out_fT, t);
  }
}
